// ABCAttention_69398081569486
// MI455X (gfx1250) — compile-verified, matrix operand order checked by reading, not run on hardware
//
#include <hip/hip_runtime.h>
#include <hip/hip_bf16.h>
#include <math.h>
#include <stdint.h>

// ---------------- problem constants (match reference) ----------------
#define B_ 2
#define T_ 2048
#define HID_ 2048
#define H_ 8
#define DK_ 128
#define DV_ 256
#define MSLOT_ 32
#define CLAMP_MIN_ (-32.0f)
#define CLAMP_MAX_ (32.0f)
#define NORM_EPS_ 1e-5f

typedef __bf16 bf16_t;
typedef __attribute__((ext_vector_type(8)))  __bf16 v8bf;
typedef __attribute__((ext_vector_type(16))) __bf16 v16bf;
typedef __attribute__((ext_vector_type(8)))  float  v8f;

// ---------------- CDNA5 async global->LDS copy (ASYNCcnt-tracked) ----------------
// 16B per lane, direct to LDS, no VGPR round trip. lds_addr = generic shared
// pointer truncated to 32 bits.
__device__ __forceinline__ void async_copy_b128(uint32_t lds_addr, const bf16_t* gptr) {
    asm volatile("global_load_async_to_lds_b128 %0, %1, off"
                 :: "v"(lds_addr), "v"(gptr)
                 : "memory");
}
__device__ __forceinline__ void wait_async4() {
    asm volatile("s_wait_asynccnt 0x4" ::: "memory");
}
__device__ __forceinline__ void wait_async0() {
    asm volatile("s_wait_asynccnt 0x0" ::: "memory");
}

// ---------------- elementwise fp32 -> bf16 ----------------
__global__ void cvt_f32_bf16_kernel(const float* __restrict__ in,
                                    bf16_t* __restrict__ out, size_t n) {
    size_t stride = (size_t)gridDim.x * blockDim.x;
    for (size_t i = (size_t)blockIdx.x * blockDim.x + threadIdx.x; i < n; i += stride)
        out[i] = (bf16_t)in[i];
}

// ---------------- transpose + convert: W(K x N) f32 -> Wt(N x K) bf16 ----------------
__global__ void transpose_cvt_kernel(const float* __restrict__ W,
                                     bf16_t* __restrict__ Wt, int K, int N) {
    size_t total = (size_t)K * N;
    size_t stride = (size_t)gridDim.x * blockDim.x;
    for (size_t idx = (size_t)blockIdx.x * blockDim.x + threadIdx.x; idx < total; idx += stride) {
        size_t n  = idx / (size_t)K;
        size_t kk = idx - n * (size_t)K;
        Wt[idx] = (bf16_t)W[kk * (size_t)N + n];
    }
}

// ---------------- bf16 WMMA GEMM: C(MxN) = A(MxK) * Bt(NxK)^T ----------------
// Block tile 128x128, 256 threads = 8 waves (wave32), wave tile 32x64 (2x4 WMMA).
// Double-buffered LDS filled by async global->LDS copies (4 per wave per tile).
#define TM 128
#define TN 128
#define TK 32

__global__ __launch_bounds__(256)
void gemm_bf16_wmma_kernel(const bf16_t* __restrict__ A,
                           const bf16_t* __restrict__ Bt,
                           float* __restrict__ C,
                           int Mdim, int Ndim, int Kdim,
                           int doClamp, float clampLo, float clampHi) {
    __shared__ __align__(16) bf16_t As[2][TM][TK];
    __shared__ __align__(16) bf16_t Bs[2][TN][TK];

    const int tid  = threadIdx.x;
    const int lane = tid & 31;
    const int wave = tid >> 5;
    const int wm   = wave & 3;   // 4 bands of 32 rows
    const int wn   = wave >> 2;  // 2 bands of 64 cols
    const int rowBase = blockIdx.y * TM;
    const int colBase = blockIdx.x * TN;

    // Staging: each thread owns two 16B chunks of A and of B per tile:
    // rows r0 and r0+64, K-chunk kc (compile-time trip count, fully unrolled).
    const int r0 = tid >> 2;
    const int kc = (tid & 3) * 8;
    const bf16_t* aP0 = A  + (size_t)(rowBase + r0) * Kdim + kc;
    const bf16_t* aP1 = aP0 + (size_t)64 * Kdim;
    const bf16_t* bP0 = Bt + (size_t)(colBase + r0) * Kdim + kc;
    const bf16_t* bP1 = bP0 + (size_t)64 * Kdim;

    uint32_t lA0[2], lA1[2], lB0[2], lB1[2];
    #pragma unroll
    for (int bi = 0; bi < 2; ++bi) {
        lA0[bi] = (uint32_t)(size_t)&As[bi][r0][kc];
        lA1[bi] = (uint32_t)(size_t)&As[bi][r0 + 64][kc];
        lB0[bi] = (uint32_t)(size_t)&Bs[bi][r0][kc];
        lB1[bi] = (uint32_t)(size_t)&Bs[bi][r0 + 64][kc];
    }

    v8f acc[2][4] = {};

    // WMMA 16-bit operand lane mapping:
    // lanes 0-15: row=lane,    K chunks {0..7, 16..23}
    // lanes 16-31: row=lane-16, K chunks {8..15, 24..31}
    const int lrow = lane & 15;
    const int lk   = (lane >> 4) * 8;

    // prologue: tile 0 -> buffer 0
    async_copy_b128(lA0[0], aP0);
    async_copy_b128(lA1[0], aP1);
    async_copy_b128(lB0[0], bP0);
    async_copy_b128(lB1[0], bP1);

    int buf = 0;
    for (int k0 = 0; k0 < Kdim; k0 += TK) {
        const int nbuf = buf ^ 1;
        if (k0 + TK < Kdim) {
            // issue next tile into the other buffer, then wait for current tile
            async_copy_b128(lA0[nbuf], aP0 + k0 + TK);
            async_copy_b128(lA1[nbuf], aP1 + k0 + TK);
            async_copy_b128(lB0[nbuf], bP0 + k0 + TK);
            async_copy_b128(lB1[nbuf], bP1 + k0 + TK);
            wait_async4();   // in-order ASYNCcnt: current tile's 4 ops complete
        } else {
            wait_async0();
        }
        __syncthreads();

        v16bf afrag[2], bfrag[4];
        #pragma unroll
        for (int mt = 0; mt < 2; ++mt) {
            int r = wm * 32 + mt * 16 + lrow;
            v8bf lo = *(const v8bf*)&As[buf][r][lk];
            v8bf hi = *(const v8bf*)&As[buf][r][lk + 16];
            #pragma unroll
            for (int i = 0; i < 8; ++i) { afrag[mt][i] = lo[i]; afrag[mt][i + 8] = hi[i]; }
        }
        #pragma unroll
        for (int nt = 0; nt < 4; ++nt) {
            int r = wn * 64 + nt * 16 + lrow;
            v8bf lo = *(const v8bf*)&Bs[buf][r][lk];
            v8bf hi = *(const v8bf*)&Bs[buf][r][lk + 16];
            #pragma unroll
            for (int i = 0; i < 8; ++i) { bfrag[nt][i] = lo[i]; bfrag[nt][i + 8] = hi[i]; }
        }
        #pragma unroll
        for (int mt = 0; mt < 2; ++mt)
            #pragma unroll
            for (int nt = 0; nt < 4; ++nt)
                acc[mt][nt] = __builtin_amdgcn_wmma_f32_16x16x32_bf16(
                    false, afrag[mt], false, bfrag[nt],
                    (short)0, acc[mt][nt], false, false);

        __syncthreads();   // all reads done before this buffer is refilled
        buf = nbuf;
    }

    // epilogue: D layout -> lanes 0-15: M=r, N=lane ; lanes 16-31: M=r+8, N=lane-16
    const int mLane = (lane >> 4) * 8;
    const int nLane = lane & 15;
    #pragma unroll
    for (int mt = 0; mt < 2; ++mt)
        #pragma unroll
        for (int nt = 0; nt < 4; ++nt)
            #pragma unroll
            for (int r = 0; r < 8; ++r) {
                int m = rowBase + wm * 32 + mt * 16 + mLane + r;
                int n = colBase + wn * 64 + nt * 16 + nLane;
                float val = acc[mt][nt][r];
                if (doClamp) val = fminf(fmaxf(val, clampLo), clampHi);
                C[(size_t)m * Ndim + n] = val;
            }
}

// ---------------- sequential ABC scan: one workgroup per (b,h) ----------------
// State in registers: hk 128x32 (16 floats/thread), hv 32x256 (32 floats/thread).
__global__ __launch_bounds__(256)
void abc_scan_kernel(const float* __restrict__ q, const float* __restrict__ k,
                     const float* __restrict__ v, const float* __restrict__ s,
                     float* __restrict__ o) {
    const int bh = blockIdx.x;
    const int b  = bh / H_;
    const int h  = bh % H_;
    const int tid = threadIdx.x;

    __shared__ float pbuf[MSLOT_], gebuf[MSLOT_], qvbuf[MSLOT_];
    __shared__ float qbuf[DK_], kbuf[DK_];
    __shared__ float partial[256];

    float hk[16];
    float hv[32];
    #pragma unroll
    for (int i = 0; i < 16; ++i) hk[i] = 0.f;
    #pragma unroll
    for (int i = 0; i < 32; ++i) hv[i] = 0.f;

    const int m_id = tid & 31;        // slot handled in pass-1
    const int dkb  = (tid >> 5) * 16; // 16-wide DK strip
    float z = 0.f;                    // running logsumexp (lanes 0..31 only)
    const float scale = rsqrtf((float)DK_);

    for (int t = 0; t < T_; ++t) {
        const size_t row = (size_t)(b * T_ + t) * H_ + h;
        if (tid < MSLOT_) {
            float sv = s[row * MSLOT_ + tid];
            float znew, ge;
            if (t == 0) { znew = sv; ge = 1.f; }
            else {
                float mx = fmaxf(z, sv);
                znew = mx + __logf(__expf(z - mx) + __expf(sv - mx));
                ge = __expf(z - znew);
            }
            float p = __expf(sv - znew);
            z = znew;
            gebuf[tid] = ge;
            pbuf[tid]  = p;
        }
        if (tid < DK_) {
            qbuf[tid] = q[row * DK_ + tid] * scale;
            kbuf[tid] = k[row * DK_ + tid];
        }
        float vv = v[row * DV_ + tid];
        __syncthreads();

        // pass 1: hk = hk*ge + k p^T ; ok[m] = q . hk[:,m]
        float ge = gebuf[m_id];
        float p  = pbuf[m_id];
        float accm = 0.f;
        #pragma unroll
        for (int i = 0; i < 16; ++i) {
            hk[i] = __builtin_fmaf(hk[i], ge, kbuf[dkb + i] * p);
            accm  = __builtin_fmaf(qbuf[dkb + i], hk[i], accm);
        }
        partial[tid] = accm;
        __syncthreads();

        if (tid < MSLOT_) {  // wave 0 exactly (wave32)
            float ok = 0.f;
            #pragma unroll
            for (int j = 0; j < 8; ++j) ok += partial[j * 32 + tid];
            float mx = ok;
            #pragma unroll
            for (int off = 16; off; off >>= 1) mx = fmaxf(mx, __shfl_xor(mx, off, 32));
            float e  = __expf(ok - mx);
            float se = e;
            #pragma unroll
            for (int off = 16; off; off >>= 1) se += __shfl_xor(se, off, 32);
            qvbuf[tid] = e / se;
        }
        __syncthreads();

        // pass 2: hv = hv*ge + p v^T ; ov[n] = qv . hv[:,n]
        float acco = 0.f;
        #pragma unroll
        for (int mm = 0; mm < 32; ++mm) {
            hv[mm] = __builtin_fmaf(hv[mm], gebuf[mm], pbuf[mm] * vv);
            acco   = __builtin_fmaf(qvbuf[mm], hv[mm], acco);
        }
        o[row * DV_ + tid] = acco;
        __syncthreads();
    }
}

// ---------------- fused RMSNorm * gw * g * sigmoid(g) -> bf16 ----------------
__global__ __launch_bounds__(256)
void norm_gate_kernel(const float* __restrict__ o, const float* __restrict__ g,
                      const float* __restrict__ gw, bf16_t* __restrict__ ob) {
    const size_t row = blockIdx.x;   // B*T*H rows of DV
    const int tid = threadIdx.x;     // 0..255 == DV
    float ov = o[row * DV_ + tid];
    float sq = ov * ov;
    __shared__ float red[8];
    #pragma unroll
    for (int off = 16; off; off >>= 1) sq += __shfl_xor(sq, off, 32);
    if ((tid & 31) == 0) red[tid >> 5] = sq;
    __syncthreads();
    float tot = 0.f;
    #pragma unroll
    for (int i = 0; i < 8; ++i) tot += red[i];
    float r  = rsqrtf(tot * (1.0f / DV_) + NORM_EPS_);
    float gv = g[row * DV_ + tid];
    float sig = 1.f / (1.f + __expf(-gv));
    ob[row * DV_ + tid] = (bf16_t)(ov * r * gw[tid] * gv * sig);
}

// ---------------- host orchestration ----------------
extern "C" void kernel_launch(void* const* d_in, const int* in_sizes, int n_in,
                              void* d_out, int out_size, void* d_ws, size_t ws_size,
                              hipStream_t stream) {
    (void)in_sizes; (void)n_in; (void)out_size; (void)ws_size;
    const float* x    = (const float*)d_in[0];
    const float* w_q  = (const float*)d_in[1];
    const float* w_k  = (const float*)d_in[2];
    const float* w_v  = (const float*)d_in[3];
    const float* w_g  = (const float*)d_in[4];
    const float* w_s  = (const float*)d_in[5];
    const float* w_o  = (const float*)d_in[6];
    const float* gnw  = (const float*)d_in[7];
    float* out = (float*)d_out;

    const int MR  = B_ * T_;        // 4096
    const int NQK = H_ * DK_;       // 1024
    const int NV  = H_ * DV_;       // 2048
    const int NS  = H_ * MSLOT_;    // 256

    char* ws = (char*)d_ws;
    size_t off = 0;
    auto carve = [&](size_t bytes) -> char* {
        off = (off + 255) & ~(size_t)255;
        char* p = ws + off;
        off += bytes;
        return p;
    };

    bf16_t* xb  = (bf16_t*)carve((size_t)MR * HID_ * sizeof(bf16_t));
    bf16_t* wqT = (bf16_t*)carve((size_t)NQK * HID_ * sizeof(bf16_t));
    bf16_t* wkT = (bf16_t*)carve((size_t)NQK * HID_ * sizeof(bf16_t));
    bf16_t* wvT = (bf16_t*)carve((size_t)NV  * HID_ * sizeof(bf16_t));
    bf16_t* wgT = (bf16_t*)carve((size_t)NV  * HID_ * sizeof(bf16_t));
    bf16_t* wsT = (bf16_t*)carve((size_t)NS  * HID_ * sizeof(bf16_t));
    bf16_t* woT = (bf16_t*)carve((size_t)HID_ * NV * sizeof(bf16_t));

    float* qb = (float*)carve((size_t)MR * NQK * sizeof(float));
    float* kb = (float*)carve((size_t)MR * NQK * sizeof(float));
    float* vb = (float*)carve((size_t)MR * NV  * sizeof(float));
    float* gb = (float*)carve((size_t)MR * NV  * sizeof(float));
    float* sb = (float*)carve((size_t)MR * NS  * sizeof(float));
    float* ob = (float*)carve((size_t)MR * NV  * sizeof(float));
    bf16_t* obh = (bf16_t*)carve((size_t)MR * NV * sizeof(bf16_t));

    // 1) convert x to bf16
    {
        size_t n = (size_t)MR * HID_;
        cvt_f32_bf16_kernel<<<4096, 256, 0, stream>>>(x, xb, n);
    }
    // 2) transpose+convert weights (K x N) -> (N x K) bf16
    transpose_cvt_kernel<<<4096, 256, 0, stream>>>(w_q, wqT, HID_, NQK);
    transpose_cvt_kernel<<<4096, 256, 0, stream>>>(w_k, wkT, HID_, NQK);
    transpose_cvt_kernel<<<4096, 256, 0, stream>>>(w_v, wvT, HID_, NV);
    transpose_cvt_kernel<<<4096, 256, 0, stream>>>(w_g, wgT, HID_, NV);
    transpose_cvt_kernel<<<4096, 256, 0, stream>>>(w_s, wsT, HID_, NS);
    transpose_cvt_kernel<<<4096, 256, 0, stream>>>(w_o, woT, NV, HID_);

    // 3) projection GEMMs (bf16 WMMA, fp32 accumulate)
    dim3 blk(256);
    gemm_bf16_wmma_kernel<<<dim3(NQK / TN, MR / TM), blk, 0, stream>>>(
        xb, wqT, qb, MR, NQK, HID_, 0, 0.f, 0.f);
    gemm_bf16_wmma_kernel<<<dim3(NQK / TN, MR / TM), blk, 0, stream>>>(
        xb, wkT, kb, MR, NQK, HID_, 0, 0.f, 0.f);
    gemm_bf16_wmma_kernel<<<dim3(NV / TN, MR / TM), blk, 0, stream>>>(
        xb, wvT, vb, MR, NV, HID_, 0, 0.f, 0.f);
    gemm_bf16_wmma_kernel<<<dim3(NV / TN, MR / TM), blk, 0, stream>>>(
        xb, wgT, gb, MR, NV, HID_, 0, 0.f, 0.f);
    gemm_bf16_wmma_kernel<<<dim3(NS / TN, MR / TM), blk, 0, stream>>>(
        xb, wsT, sb, MR, NS, HID_, 1, CLAMP_MIN_, CLAMP_MAX_);

    // 4) sequential gated two-pass scan (16 independent (b,h) streams)
    abc_scan_kernel<<<B_ * H_, 256, 0, stream>>>(qb, kb, vb, sb, ob);

    // 5) fused RMSNorm + swish gate -> bf16
    norm_gate_kernel<<<MR * H_, DV_, 0, stream>>>(ob, gb, gnw, obh);

    // 6) output projection straight into d_out (fp32)
    gemm_bf16_wmma_kernel<<<dim3(HID_ / TN, MR / TM), blk, 0, stream>>>(
        obh, woT, out, MR, HID_, NV, 0, 0.f, 0.f);
}
